// SparseBPNeuralNetwork_19825569038601
// MI455X (gfx1250) — matrix operand order repaired and checked by reading, not compiled, not run on hardware
//
#include <hip/hip_runtime.h>
#include <math.h>

// ---------------------------------------------------------------------------
// Sparse BP decoder on MI455X (gfx1250): bf16 WMMA GEMM pipeline with
// Tensor-Data-Mover (TDM) async A-tile staging and double-buffered LDS.
// ---------------------------------------------------------------------------

typedef __attribute__((ext_vector_type(16))) __bf16        v16bf;
typedef __attribute__((ext_vector_type(8)))  float         v8f;
typedef __attribute__((ext_vector_type(4)))  unsigned int  u32x4;
typedef __attribute__((ext_vector_type(2)))  unsigned int  u32x2;
typedef __attribute__((ext_vector_type(4)))  float         f32x4;
typedef __attribute__((ext_vector_type(8)))  int           i32x8;
typedef __attribute__((ext_vector_type(4)))  int           i32x4;

union FragBF { v16bf v; u32x4 q[2]; };

#define CLIPV 0.999999f

#if __has_include(<hip/amd_detail/amd_gfx1250_TDM.h>)
#define TDM_ARGS6 1
#else
#define TDM_ARGS6 0
#endif

__device__ __forceinline__ unsigned short f2bf(float f) {
  unsigned int x = __float_as_uint(f);
  x += 0x7FFFu + ((x >> 16) & 1u);          // round-to-nearest-even
  return (unsigned short)(x >> 16);
}
__device__ __forceinline__ unsigned int pk2(float lo, float hi) {
  return (unsigned int)f2bf(lo) | ((unsigned int)f2bf(hi) << 16);
}

static constexpr int BM  = 128;   // rows per workgroup tile (batch dim)
static constexpr int BN  = 64;    // cols per workgroup tile
static constexpr int BK  = 32;    // K per WMMA step
static constexpr int AST = 40;    // LDS row stride in ushort (32 data + 8 pad)

// ---------------------------------------------------------------------------
// TDM: async 2D tile load, global bf16 [rows x lda] -> LDS with padded rows.
// Tile = 128 rows x 32 elements (64B data + 16B pad per LDS row = 40 ushorts).
// D# per ISA 8.3/8.4; issued by wave 0 only (TDM ignores EXEC, one DMA/tile).
// ---------------------------------------------------------------------------
__device__ __forceinline__ void tdm_load_tile_a(unsigned lds_off,
                                                const unsigned short* gptr,
                                                int lda) {
#if defined(__AMDGCN__)
  unsigned long long ga = (unsigned long long)(size_t)gptr;
  u32x4 g0;
  g0[0] = 1u;                                            // count=1, user mode
  g0[1] = lds_off;                                       // LDS byte address
  g0[2] = (unsigned)(ga & 0xFFFFFFFFu);                  // global_addr[31:0]
  g0[3] = (unsigned)((ga >> 32) & 0x01FFFFFFu)           // global_addr[56:32]
          | 0x80000000u;                                 // type=2 ("image")
  i32x8 g1;
  // data_size=1 (2B), pad_enable, pad_interval=3 (16 DW), pad_amount=3 (4 DW)
  g1[0] = (int)0x06D10000u;
  g1[1] = (int)(((unsigned)lda & 0xFFFFu) << 16);        // tensor_dim0[15:0]
  g1[2] = (int)((((unsigned)lda >> 16) & 0xFFFFu)        // tensor_dim0[31:16]
          | (256u << 16));                               // tensor_dim1[15:0]
  g1[3] = (int)(32u << 16);                              // tile_dim0 = 32
  g1[4] = 128;                                           // tile_dim1 = 128
  g1[5] = lda;                                           // dim0_stride[31:0]
  g1[6] = 0;
  g1[7] = 0;
  i32x4 z4; z4[0] = 0; z4[1] = 0; z4[2] = 0; z4[3] = 0;  // groups 2/3 unused
#if TDM_ARGS6
  i32x8 z8;
#pragma unroll
  for (int i = 0; i < 8; ++i) z8[i] = 0;
  __builtin_amdgcn_tensor_load_to_lds(g0, g1, z4, z4, z8, 0);
#else
  __builtin_amdgcn_tensor_load_to_lds(g0, g1, z4, z4, 0);
#endif
#else
  (void)lds_off; (void)gptr; (void)lda;
#endif
}

__device__ __forceinline__ void tdm_wait0() {
#if defined(__AMDGCN__)
  __builtin_amdgcn_s_wait_tensorcnt(0);
#endif
}

__device__ __forceinline__ unsigned lds_addr_of(const void* p) {
  return (unsigned)(size_t)p;   // low 32 bits of generic LDS pointer = offset
}

// ---------------------------------------------------------------------------
// Generic GEMM: C[MxN] = A(bf16)[MxK] x B, where
//   WT=1: weight f32 row-major [N_out, K]  (computes A @ W.T)
//   WT=0: weight f32 row-major [K, N_out]  (computes A @ W)
// Epilogues:
//   EP=0: store bf16 to o1
//   EP=1: store f32 to outF
//   EP=2: h = tanh(0.5*(acc+bias)); emit CN inputs log|h|, (h<0), (h!=0) (bf16)
//   EP=3: store f32 sigmoid(acc+bias) to outF
// ---------------------------------------------------------------------------
template <int WT, int EP>
__global__ __launch_bounds__(256) void gemm_bf16(
    const unsigned short* __restrict__ A, int lda,
    const float* __restrict__ W, int ldw,
    const float* __restrict__ bias, int ldb,
    float* __restrict__ outF,
    unsigned short* __restrict__ o1,
    unsigned short* __restrict__ o2,
    unsigned short* __restrict__ o3,
    int ldo, int K)
{
  __shared__ unsigned short As[2][BM * AST];
  __shared__ unsigned short Bs[2][BN * AST];

  const int tid  = threadIdx.x;
  const int w    = tid >> 5;
  const int lane = tid & 31;
  const int half = lane >> 4;
  const int l16  = lane & 15;
  const int wm   = (w >> 1) * 32;   // 4x2 wave grid, 32x32 per wave
  const int wn   = (w & 1) * 32;
  const int m0   = blockIdx.x * BM;
  const int n0   = blockIdx.y * BN;

  f32x4 wreg[2];

  auto loadB = [&](int k0) {
#pragma unroll
    for (int it = 0; it < 2; ++it) {
      int cid = it * 256 + tid;
      if (WT) {
        int n = cid >> 3, kc = (cid & 7) * 4;
        wreg[it] = *(const f32x4*)&W[(size_t)(n0 + n) * ldw + k0 + kc];
      } else {
        int kk = cid >> 4, nc = (cid & 15) * 4;
        wreg[it] = *(const f32x4*)&W[(size_t)(k0 + kk) * ldw + n0 + nc];
      }
    }
  };
  auto storeB = [&](int buf) {
#pragma unroll
    for (int it = 0; it < 2; ++it) {
      int cid = it * 256 + tid;
      if (WT) {
        int n = cid >> 3, kc = (cid & 7) * 4;
        u32x2 pq; pq[0] = pk2(wreg[it][0], wreg[it][1]);
        pq[1] = pk2(wreg[it][2], wreg[it][3]);
        *(u32x2*)&Bs[buf][n * AST + kc] = pq;
      } else {
        int kk = cid >> 4, nc = (cid & 15) * 4;
        Bs[buf][(nc + 0) * AST + kk] = f2bf(wreg[it][0]);
        Bs[buf][(nc + 1) * AST + kk] = f2bf(wreg[it][1]);
        Bs[buf][(nc + 2) * AST + kk] = f2bf(wreg[it][2]);
        Bs[buf][(nc + 3) * AST + kk] = f2bf(wreg[it][3]);
      }
    }
  };
  auto stageA = [&](int k0, int buf) {
    if (w == 0)
      tdm_load_tile_a(lds_addr_of(&As[buf][0]),
                      &A[(size_t)m0 * lda + k0], lda);
  };

  v8f acc[2][2];
#pragma unroll
  for (int mi = 0; mi < 2; ++mi)
#pragma unroll
    for (int ni = 0; ni < 2; ++ni)
#pragma unroll
      for (int r = 0; r < 8; ++r) acc[mi][ni][r] = 0.0f;

  // prologue: stage tile 0
  stageA(0, 0);
  loadB(0);
  storeB(0);
  if (w == 0) tdm_wait0();
  __syncthreads();

  int buf = 0;
  for (int k0 = 0; k0 < K; k0 += BK) {
    const bool more = (k0 + BK < K);
    if (more) {                        // overlap: next tile in flight
      stageA(k0 + BK, buf ^ 1);        // TDM (async, TENSORcnt)
      loadB(k0 + BK);                  // weight panel -> registers
    }

    FragBF a[2], b[2];
#pragma unroll
    for (int mi = 0; mi < 2; ++mi) {
      const unsigned short* p = &As[buf][(wm + mi * 16 + l16) * AST + 8 * half];
      a[mi].q[0] = *(const u32x4*)p;          // K = 0..7  (+8*half)
      a[mi].q[1] = *(const u32x4*)(p + 16);   // K = 16..23 (+8*half)
    }
#pragma unroll
    for (int ni = 0; ni < 2; ++ni) {
      const unsigned short* p = &Bs[buf][(wn + ni * 16 + l16) * AST + 8 * half];
      b[ni].q[0] = *(const u32x4*)p;          // K = 0..7  (+8*half)
      b[ni].q[1] = *(const u32x4*)(p + 16);   // K = 16..23 (+8*half)
    }
    if (more) storeB(buf ^ 1);         // other buffer: no hazard with readers

#pragma unroll
    for (int mi = 0; mi < 2; ++mi)
#pragma unroll
      for (int ni = 0; ni < 2; ++ni)
        acc[mi][ni] = __builtin_amdgcn_wmma_f32_16x16x32_bf16(
            false, a[mi].v, false, b[ni].v, (short)0, acc[mi][ni], false, false);

    if (more) {
      if (w == 0) tdm_wait0();
      __syncthreads();
    }
    buf ^= 1;
  }

  // ---- epilogue: element (M = r + 8*half, N = l16) per C/D layout ----
#pragma unroll
  for (int mi = 0; mi < 2; ++mi)
#pragma unroll
    for (int ni = 0; ni < 2; ++ni)
#pragma unroll
      for (int r = 0; r < 8; ++r) {
        int gr = m0 + wm + mi * 16 + r + 8 * half;
        int gc = n0 + wn + ni * 16 + l16;
        float vacc = acc[mi][ni][r];
        if (EP == 0) {
          o1[(size_t)gr * ldo + gc] = f2bf(vacc);
        } else if (EP == 1) {
          outF[(size_t)gr * ldo + gc] = vacc;
        } else if (EP == 2) {
          float h = tanhf(0.5f * (vacc + bias[(size_t)gr * ldb + gc]));
          o1[(size_t)gr * ldo + gc] = f2bf(h != 0.0f ? logf(fabsf(h)) : 0.0f);
          o2[(size_t)gr * ldo + gc] = f2bf(h < 0.0f ? 1.0f : 0.0f);
          o3[(size_t)gr * ldo + gc] = f2bf(h != 0.0f ? 1.0f : 0.0f);
        } else {
          float s = vacc + bias[(size_t)gr * ldb + gc];
          outF[(size_t)gr * ldo + gc] = 1.0f / (1.0f + expf(-s));
        }
      }
}

// ---------------------------------------------------------------------------
// CN update: three GEMMs (log|h|, negcount, nzcount) sharing one B-tile of the
// mask M (f32 [N_out, K], A@M.T). Three TDM loads/tile, 12 WMMAs/K-step.
// Epilogue: masked product with zero handling, clip, 2*atanh -> bf16.
// ---------------------------------------------------------------------------
__global__ __launch_bounds__(256) void cn3_bf16(
    const unsigned short* __restrict__ AL,
    const unsigned short* __restrict__ AN,
    const unsigned short* __restrict__ AC,
    int lda,
    const float* __restrict__ W, int ldw,
    unsigned short* __restrict__ outT, int ldo, int K)
{
  __shared__ unsigned short AsL[2][BM * AST];
  __shared__ unsigned short AsN[2][BM * AST];
  __shared__ unsigned short AsC[2][BM * AST];
  __shared__ unsigned short Bs[2][BN * AST];

  const int tid  = threadIdx.x;
  const int w    = tid >> 5;
  const int lane = tid & 31;
  const int half = lane >> 4;
  const int l16  = lane & 15;
  const int wm   = (w >> 1) * 32;
  const int wn   = (w & 1) * 32;
  const int m0   = blockIdx.x * BM;
  const int n0   = blockIdx.y * BN;

  f32x4 wreg[2];

  auto loadB = [&](int k0) {
#pragma unroll
    for (int it = 0; it < 2; ++it) {
      int cid = it * 256 + tid;
      int n = cid >> 3, kc = (cid & 7) * 4;
      wreg[it] = *(const f32x4*)&W[(size_t)(n0 + n) * ldw + k0 + kc];
    }
  };
  auto storeB = [&](int buf) {
#pragma unroll
    for (int it = 0; it < 2; ++it) {
      int cid = it * 256 + tid;
      int n = cid >> 3, kc = (cid & 7) * 4;
      u32x2 pq; pq[0] = pk2(wreg[it][0], wreg[it][1]);
      pq[1] = pk2(wreg[it][2], wreg[it][3]);
      *(u32x2*)&Bs[buf][n * AST + kc] = pq;
    }
  };
  auto stageA = [&](int k0, int buf) {
    if (w == 0) {
      size_t g = (size_t)m0 * lda + k0;
      tdm_load_tile_a(lds_addr_of(&AsL[buf][0]), &AL[g], lda);
      tdm_load_tile_a(lds_addr_of(&AsN[buf][0]), &AN[g], lda);
      tdm_load_tile_a(lds_addr_of(&AsC[buf][0]), &AC[g], lda);
    }
  };

  v8f accL[2][2], accN[2][2], accC[2][2];
#pragma unroll
  for (int mi = 0; mi < 2; ++mi)
#pragma unroll
    for (int ni = 0; ni < 2; ++ni)
#pragma unroll
      for (int r = 0; r < 8; ++r) {
        accL[mi][ni][r] = 0.0f; accN[mi][ni][r] = 0.0f; accC[mi][ni][r] = 0.0f;
      }

  stageA(0, 0);
  loadB(0);
  storeB(0);
  if (w == 0) tdm_wait0();
  __syncthreads();

  int buf = 0;
  for (int k0 = 0; k0 < K; k0 += BK) {
    const bool more = (k0 + BK < K);
    if (more) { stageA(k0 + BK, buf ^ 1); loadB(k0 + BK); }

    FragBF aL[2], aN[2], aC[2], b[2];
#pragma unroll
    for (int mi = 0; mi < 2; ++mi) {
      int row = (wm + mi * 16 + l16) * AST + 8 * half;
      aL[mi].q[0] = *(const u32x4*)&AsL[buf][row];
      aL[mi].q[1] = *(const u32x4*)&AsL[buf][row + 16];
      aN[mi].q[0] = *(const u32x4*)&AsN[buf][row];
      aN[mi].q[1] = *(const u32x4*)&AsN[buf][row + 16];
      aC[mi].q[0] = *(const u32x4*)&AsC[buf][row];
      aC[mi].q[1] = *(const u32x4*)&AsC[buf][row + 16];
    }
#pragma unroll
    for (int ni = 0; ni < 2; ++ni) {
      const unsigned short* p = &Bs[buf][(wn + ni * 16 + l16) * AST + 8 * half];
      b[ni].q[0] = *(const u32x4*)p;
      b[ni].q[1] = *(const u32x4*)(p + 16);
    }
    if (more) storeB(buf ^ 1);

#pragma unroll
    for (int mi = 0; mi < 2; ++mi)
#pragma unroll
      for (int ni = 0; ni < 2; ++ni) {
        accL[mi][ni] = __builtin_amdgcn_wmma_f32_16x16x32_bf16(
            false, aL[mi].v, false, b[ni].v, (short)0, accL[mi][ni], false, false);
        accN[mi][ni] = __builtin_amdgcn_wmma_f32_16x16x32_bf16(
            false, aN[mi].v, false, b[ni].v, (short)0, accN[mi][ni], false, false);
        accC[mi][ni] = __builtin_amdgcn_wmma_f32_16x16x32_bf16(
            false, aC[mi].v, false, b[ni].v, (short)0, accC[mi][ni], false, false);
      }

    if (more) {
      if (w == 0) tdm_wait0();
      __syncthreads();
    }
    buf ^= 1;
  }

#pragma unroll
  for (int mi = 0; mi < 2; ++mi)
#pragma unroll
    for (int ni = 0; ni < 2; ++ni)
#pragma unroll
      for (int r = 0; r < 8; ++r) {
        int gr = m0 + wm + mi * 16 + r + 8 * half;
        int gc = n0 + wn + ni * 16 + l16;
        float sl = accL[mi][ni][r];
        float sn = accN[mi][ni][r];
        float sc = accC[mi][ni][r];
        float m    = sn - 2.0f * floorf(sn * 0.5f);       // negc % 2
        float prod = expf(sl) * (1.0f - 2.0f * m);
        float res  = (sc > 0.5f) ? prod : 0.0f;           // cnt > 0
        res = fminf(fmaxf(res, -CLIPV), CLIPV);
        outT[(size_t)gr * ldo + gc] = f2bf(logf((1.0f + res) / (1.0f - res)));
      }
}

// Initial prep: llr bf16, and CN inputs for tanh(0.5*x).
__global__ void prep_kernel(const float* __restrict__ x,
                            unsigned short* __restrict__ llr,
                            unsigned short* __restrict__ alog,
                            unsigned short* __restrict__ aneg,
                            unsigned short* __restrict__ anz, int n)
{
  int i = blockIdx.x * blockDim.x + threadIdx.x;
  if (i < n) {
    float v = x[i];
    llr[i] = f2bf(v);
    float h = tanhf(0.5f * v);
    alog[i] = f2bf(h != 0.0f ? logf(fabsf(h)) : 0.0f);
    aneg[i] = f2bf(h < 0.0f ? 1.0f : 0.0f);
    anz[i]  = f2bf(h != 0.0f ? 1.0f : 0.0f);
  }
}

extern "C" void kernel_launch(void* const* d_in, const int* in_sizes, int n_in,
                              void* d_out, int out_size, void* d_ws, size_t ws_size,
                              hipStream_t stream)
{
  (void)in_sizes; (void)n_in; (void)out_size; (void)ws_size;
  const int B = 256, N = 768, H = 3072;

  const float* x    = (const float*)d_in[0];
  const float* W_vn = (const float*)d_in[1];   // [19, H, H]
  const float* W_o  = (const float*)d_in[2];   // [N, H]
  const float* S    = (const float*)d_in[3];   // [20, N, N]
  const float* bm   = (const float*)d_in[4];   // [N, H]
  const float* cm   = (const float*)d_in[5];   // [N, N]
  const float* Mf   = (const float*)d_in[6];   // [H, N]
  const float* Mc   = (const float*)d_in[7];   // [H, H]

  char* p = (char*)d_ws;
  auto alloc = [&](size_t bytes) -> void* {
    void* r = (void*)p; p += (bytes + 255) & ~(size_t)255; return r;
  };
  unsigned short* tb    = (unsigned short*)alloc((size_t)B * H * 2);  // bf16 act
  unsigned short* alog  = (unsigned short*)alloc((size_t)B * H * 2);
  unsigned short* aneg  = (unsigned short*)alloc((size_t)B * H * 2);
  unsigned short* anz   = (unsigned short*)alloc((size_t)B * H * 2);
  unsigned short* llr   = (unsigned short*)alloc((size_t)B * N * 2);
  unsigned short* t2    = (unsigned short*)alloc((size_t)B * N * 2);
  float*          bias2 = (float*)alloc((size_t)B * H * 4);
  float*          bias3 = (float*)alloc((size_t)B * N * 4);

  dim3 blk(256);
  prep_kernel<<<(B * N + 255) / 256, 256, 0, stream>>>(x, llr, alog, aneg, anz, B * N);

  // fc0: CN update of tanh(0.5*x) with M_first  ->  t = 2*atanh(clip(.)) bf16 [B,H]
  cn3_bf16<<<dim3(B / BM, H / BN), blk, 0, stream>>>(alog, aneg, anz, N, Mf, N, tb, H, N);

  for (int i = 0; i < 19; ++i) {
    // t2 = llr @ S_i                       (bf16 out)
    gemm_bf16<0, 0><<<dim3(B / BM, N / BN), blk, 0, stream>>>(
        llr, N, S + (size_t)i * N * N, N, nullptr, 0, nullptr, t2, nullptr, nullptr, N, N);
    // bias2 = t2 @ bias_matrix             (f32 out [B,H])
    gemm_bf16<0, 1><<<dim3(B / BM, H / BN), blk, 0, stream>>>(
        t2, N, bm, H, nullptr, 0, bias2, nullptr, nullptr, nullptr, H, N);
    // VN: h = tanh(0.5*(t @ W_i.T + bias2)) -> emit CN inputs
    gemm_bf16<1, 2><<<dim3(B / BM, H / BN), blk, 0, stream>>>(
        tb, H, W_vn + (size_t)i * H * H, H, bias2, H, nullptr, alog, aneg, anz, H, H);
    // CN update with M_cn -> next t
    cn3_bf16<<<dim3(B / BM, H / BN), blk, 0, stream>>>(alog, aneg, anz, H, Mc, H, tb, H, H);
  }

  // Output stage
  gemm_bf16<0, 0><<<dim3(B / BM, N / BN), blk, 0, stream>>>(
      llr, N, S + (size_t)19 * N * N, N, nullptr, 0, nullptr, t2, nullptr, nullptr, N, N);
  gemm_bf16<0, 1><<<dim3(B / BM, N / BN), blk, 0, stream>>>(
      t2, N, cm, N, nullptr, 0, bias3, nullptr, nullptr, nullptr, N, N);
  gemm_bf16<1, 3><<<dim3(B / BM, N / BN), blk, 0, stream>>>(
      tb, H, W_o, H, bias3, N, (float*)d_out, nullptr, nullptr, nullptr, N, H);
}
